// LTAE2dtiny_76038101008657
// MI455X (gfx1250) — matrix operand order repaired and checked by reading, not compiled, not run on hardware
//
#include <hip/hip_runtime.h>
#include <stdint.h>

#define IN_CH   128
#define N_HEAD  16
#define D_MODEL 256
#define L_SEQ   30
#define SZB     4
#define HGT     64
#define WID     64
#define PIX     8            // pixels per workgroup (contiguous along x)
#define T_PE    1000.0f

typedef _Float16 v16h  __attribute__((ext_vector_type(16)));
typedef float    v8f   __attribute__((ext_vector_type(8)));
typedef uint32_t u32x4 __attribute__((ext_vector_type(4)));
typedef uint32_t u32x8 __attribute__((ext_vector_type(8)));

// ---------------------------------------------------------------------------
// Fold kernel 1: W2q[h,c] = sum_d Q[h,d] * (Wk @ conv_w)[4h+d, c]
//   Bmat[h,c]  = f16( gamma[c] * W2q[h,c] )   (WMMA B operand, constant)
// ---------------------------------------------------------------------------
__global__ __launch_bounds__(256) void fold_w2q(
    const float* __restrict__ Q, const float* __restrict__ Wk,
    const float* __restrict__ conv_w, const float* __restrict__ gamma,
    float* __restrict__ W2qf, _Float16* __restrict__ Bmat)
{
    int t = blockIdx.x * blockDim.x + threadIdx.x;
    if (t >= N_HEAD * IN_CH) return;
    int c  = t & (IN_CH - 1);
    int hh = t >> 7;
    float q0 = Q[hh*4+0], q1 = Q[hh*4+1], q2 = Q[hh*4+2], q3 = Q[hh*4+3];
    float acc = 0.f;
    for (int m = 0; m < D_MODEL; ++m) {
        float qk = q0 * Wk[(hh*4+0)*D_MODEL + m] + q1 * Wk[(hh*4+1)*D_MODEL + m]
                 + q2 * Wk[(hh*4+2)*D_MODEL + m] + q3 * Wk[(hh*4+3)*D_MODEL + m];
        acc += qk * conv_w[m*IN_CH + c];
    }
    W2qf[hh*IN_CH + c] = acc;
    Bmat[hh*IN_CH + c] = (_Float16)(gamma[c] * acc);
}

// ---------------------------------------------------------------------------
// Fold kernel 2: cfin[sb,l,h] = 0.5*( hbias[h] + peScore[sb,l,h] )
//   hbias[h]  = sum_c beta[c]*W2q[h,c] + sum_d Q[h,d]*(Wk[4h+d,:]@conv_b + bk)
//   peScore   = sum_d Q[h,d]* sum_m Wk[4h+d,m]*pe_base[sb,l,m%16]
// ---------------------------------------------------------------------------
__global__ __launch_bounds__(256) void fold_const(
    const float* __restrict__ Q, const float* __restrict__ Wk,
    const float* __restrict__ conv_b, const float* __restrict__ bk,
    const float* __restrict__ beta, const int* __restrict__ bpos,
    const float* __restrict__ W2qf, float* __restrict__ cfin)
{
    int t = blockIdx.x * blockDim.x + threadIdx.x;
    if (t >= SZB * L_SEQ * N_HEAD) return;
    int hh = t & 15;
    int l  = (t >> 4) % L_SEQ;
    int sb = t / (L_SEQ * N_HEAD);

    float hb = 0.f;
    for (int c = 0; c < IN_CH; ++c) hb += beta[c] * W2qf[hh*IN_CH + c];
    for (int d = 0; d < 4; ++d) {
        float dot = 0.f;
        for (int m = 0; m < D_MODEL; ++m) dot += Wk[(hh*4+d)*D_MODEL + m] * conv_b[m];
        hb += Q[hh*4+d] * (dot + bk[hh*4+d]);
    }
    float pos = (float)bpos[sb*L_SEQ + l];
    float base[16];
    for (int j = 0; j < 16; ++j) {
        float denom = powf(T_PE, (2.0f * (float)(j >> 1)) / 16.0f);
        float a = pos / denom;
        base[j] = (j & 1) ? cosf(a) : sinf(a);
    }
    float pes = 0.f;
    for (int d = 0; d < 4; ++d) {
        float dot = 0.f;
        for (int m = 0; m < D_MODEL; ++m) dot += Wk[(hh*4+d)*D_MODEL + m] * base[m & 15];
        pes += Q[hh*4+d] * dot;
    }
    cfin[t] = 0.5f * (hb + pes);
}

// ---------------------------------------------------------------------------
// Main kernel: TDM tile gather -> GN stats -> f16 WMMA scores -> softmax
// ---------------------------------------------------------------------------
#define XS_WORDS     (L_SEQ * IN_CH * PIX)          /* 30720 */
#define XS_WORDS_PAD (XS_WORDS + (XS_WORDS >> 8))   /* +1 dw per 256 (TDM pad) */

__device__ __forceinline__ int xsw(int p) { return p + (p >> 8); }

__global__ __launch_bounds__(256) void ltae_main(
    const float* __restrict__ x, const unsigned char* __restrict__ pad_mask,
    const _Float16* __restrict__ Bmat, const float* __restrict__ cfin,
    float* __restrict__ out)
{
    extern __shared__ float smem[];
    float* xs    = smem;                     // [l][c][px] (TDM-padded)
    float* spart = smem + XS_WORDS_PAD;      // 512
    float* mus   = spart + 512;              // 128 : [g*8+px]
    float* invs  = mus + 128;                // 128
    float* outs  = invs + 128;               // 16*30*8 = 3840

    const int t   = threadIdx.x;
    const int blk = blockIdx.x;              // 2048 = 4 b * 64 y * 8 xg
    const int b   = blk >> 9;
    const int rem = blk & 511;
    const int y   = rem >> 3;
    const int x0  = (rem & 7) * PIX;

    // ---- TDM: one DMA pulls the [8px,128c,30l] tile (strided gather) -------
    if (t < 32) {
        uint64_t gaddr = (uint64_t)(uintptr_t)x +
            4ull * ((uint64_t)b * L_SEQ * IN_CH * HGT * WID + (uint64_t)y * WID + x0);
        uint32_t lds0 = (uint32_t)(uintptr_t)xs;   // low 32 bits == LDS byte offset
        u32x4 g0; u32x8 g1; u32x4 g2; u32x4 g3;
        g0[0] = 1u;                                          // count=1, user mode
        g0[1] = lds0;                                        // lds_addr
        g0[2] = (uint32_t)gaddr;                             // global_addr[31:0]
        g0[3] = (uint32_t)((gaddr >> 32) & 0x1FFFFFFu) | (2u << 30); // addr hi | type=2
        g1[0] = (2u << 16) | (1u << 20) | (7u << 22);        // 4B elems, pad 1dw/256dw
        g1[1] = 0u;                                          // tensor_dim0 = 1<<20 (lo16=0)
        g1[2] = 0x10u;                                       // dim0 hi | dim1 lo (1<<20)
        g1[3] = 0x10u | (8u << 16);                          // dim1 hi | tile_dim0=8 (px)
        g1[4] = 128u | (30u << 16);                          // tile_dim1=128(c), tile_dim2=30(l)
        g1[5] = 4096u;                                       // dim0_stride = h*w (c step)
        g1[6] = 0u;                                          // dim0_stride hi | dim1_stride lo
        g1[7] = 8u;                                          // dim1_stride = 524288 (l step) >>16
        g2[0] = (1u << 20); g2[1] = 1u; g2[2] = 0u; g2[3] = 0u;
        g3[0] = 0u; g3[1] = 0u; g3[2] = 0u; g3[3] = 0u;
        asm volatile("tensor_load_to_lds %0, %1, %2, %3"
                     :: "s"(g0), "s"(g1), "s"(g2), "s"(g3) : "memory");
        __builtin_amdgcn_s_wait_tensorcnt(0);
    }
    __syncthreads();

    // ---- GroupNorm stats: per (pixel, head-group) over 8 ch x 30 steps -----
    {
        int spx = t & 7, g = (t >> 3) & 15, rep = t >> 7;
        float s1 = 0.f, s2 = 0.f;
        for (int li = 0; li < 15; ++li) {
            int l = rep * 15 + li;
            int base = l * 1024 + g * 64 + spx;
            #pragma unroll
            for (int cc = 0; cc < 8; ++cc) {
                float v = xs[xsw(base + cc * 8)];
                s1 += v; s2 += v * v;
            }
        }
        spart[t] = s1; spart[256 + t] = s2;
    }
    __syncthreads();
    if (t < 128) {
        float s1 = spart[t] + spart[t + 128];
        float s2 = spart[256 + t] + spart[256 + t + 128];
        float mu  = s1 * (1.0f / 240.0f);
        float var = s2 * (1.0f / 240.0f) - mu * mu;
        mus[t]  = mu;
        invs[t] = rsqrtf(var + 1e-5f);
    }
    __syncthreads();

    // ---- WMMA: per wave, one pixel: [32(l) x 128(c)] * [128(c) x 16(h)] ----
    const int lane = t & 31;
    const int px   = t >> 5;                 // wave id == pixel
    const int hh   = lane & 15;
    const int hi   = lane >> 4;

    v16h bfr[4];                             // constant B fragments (f16)
    #pragma unroll
    for (int kk = 0; kk < 4; ++kk)
        #pragma unroll
        for (int r = 0; r < 8; ++r) {
            int c = kk * 32 + hi * 16 + 2 * r;
            bfr[kk][2*r]   = Bmat[hh * IN_CH + c];
            bfr[kk][2*r+1] = Bmat[hh * IN_CH + c + 1];
        }

    v8f acc[2] = {};
    #pragma unroll
    for (int tile = 0; tile < 2; ++tile) {
        const int l = tile * 16 + (lane & 15);       // A-matrix row
        #pragma unroll
        for (int kk = 0; kk < 4; ++kk) {
            v16h a;
            #pragma unroll
            for (int i = 0; i < 16; ++i) a[i] = (_Float16)0.0f;
            if (l < L_SEQ) {
                #pragma unroll
                for (int r = 0; r < 8; ++r) {
                    int k = 2 * r + hi * 16;
                    int c = kk * 32 + k;
                    int g = c >> 3;
                    float mu = mus[g * 8 + px], iv = invs[g * 8 + px];
                    int p0 = l * 1024 + c * 8 + px;
                    a[2*r]   = (_Float16)((xs[xsw(p0)]     - mu) * iv);
                    a[2*r+1] = (_Float16)((xs[xsw(p0 + 8)] - mu) * iv);
                }
            }
            acc[tile] = __builtin_amdgcn_wmma_f32_16x16x32_f16(
                false, a, false, bfr[kk], (short)0, acc[tile], false, false);
        }
    }

    // ---- softmax over L (head = lane%16; partner lane^16 holds other l's) --
    float sc[16];
    float mx = -3.0e38f;
    #pragma unroll
    for (int tile = 0; tile < 2; ++tile)
        #pragma unroll
        for (int j = 0; j < 8; ++j) {
            int l = tile * 16 + hi * 8 + j;          // D row for this lane/VGPR
            float s = -3.0e38f;
            if (l < L_SEQ) {
                s = 0.5f * acc[tile][j] + cfin[(b * L_SEQ + l) * N_HEAD + hh];
                if (pad_mask[b * L_SEQ + l]) s = -1000.0f;
            }
            sc[tile * 8 + j] = s;
            mx = fmaxf(mx, s);
        }
    mx = fmaxf(mx, __shfl_xor(mx, 16, 32));
    float sum = 0.f;
    #pragma unroll
    for (int i = 0; i < 16; ++i) { sc[i] = __expf(sc[i] - mx); sum += sc[i]; }
    sum += __shfl_xor(sum, 16, 32);
    float rs = 1.0f / sum;
    #pragma unroll
    for (int tile = 0; tile < 2; ++tile)
        #pragma unroll
        for (int j = 0; j < 8; ++j) {
            int l = tile * 16 + hi * 8 + j;
            if (l < L_SEQ) outs[(hh * L_SEQ + l) * PIX + px] = sc[tile * 8 + j] * rs;
        }
    __syncthreads();

    // ---- coalesced store: out[h][b][l][y][x0..x0+7] ------------------------
    for (int i = t; i < N_HEAD * L_SEQ * PIX; i += 256) {
        int pxo  = i & 7;
        int rest = i >> 3;
        int l    = rest % L_SEQ;
        int hhh  = rest / L_SEQ;
        out[(((hhh * SZB + b) * L_SEQ + l) * HGT + y) * WID + x0 + pxo] = outs[i];
    }
}

// ---------------------------------------------------------------------------
extern "C" void kernel_launch(void* const* d_in, const int* in_sizes, int n_in,
                              void* d_out, int out_size, void* d_ws, size_t ws_size,
                              hipStream_t stream)
{
    (void)in_sizes; (void)n_in; (void)out_size; (void)ws_size;
    const float* x     = (const float*)d_in[0];
    const float* gnw   = (const float*)d_in[1];
    const float* gnb   = (const float*)d_in[2];
    const float* convw = (const float*)d_in[3];
    const float* convb = (const float*)d_in[4];
    const float* Q     = (const float*)d_in[5];
    const float* Wk    = (const float*)d_in[6];
    const float* bk    = (const float*)d_in[7];
    const int*   bpos  = (const int*)d_in[8];
    const unsigned char* pmask = (const unsigned char*)d_in[9];
    float* out = (float*)d_out;

    float*    W2qf = (float*)d_ws;                              // 2048 f32
    _Float16* Bmat = (_Float16*)((char*)d_ws + 8192);           // 2048 f16
    float*    cfin = (float*)((char*)d_ws + 12288);             // 1920 f32

    fold_w2q  <<<8, 256, 0, stream>>>(Q, Wk, convw, gnw, W2qf, Bmat);
    fold_const<<<8, 256, 0, stream>>>(Q, Wk, convb, bk, gnb, bpos, W2qf, cfin);

    constexpr size_t LDS_BYTES =
        (size_t)(XS_WORDS_PAD + 512 + 128 + 128 + N_HEAD * L_SEQ * PIX) * 4;
    ltae_main<<<SZB * HGT * (WID / PIX), 256, LDS_BYTES, stream>>>(
        x, pmask, Bmat, cfin, out);
}
